// TAttention_76785425318728
// MI455X (gfx1250) — matrix operand order repaired and checked by reading, not compiled, not run on hardware
//
#include <hip/hip_runtime.h>
#include <hip/hip_bf16.h>

// ---------------------------------------------------------------------------
// GPT-2 style transformer block for MI455X (gfx1250, wave32, WMMA + TDM).
// GEMMs + attention use v_wmma_f32_16x16x32_bf16 (f32 accumulate).
// A-tiles staged by the Tensor Data Mover (double-buffered, TENSORcnt);
// B-tiles staged transposed in LDS so both fragment reads are ds_load_b128.
// ---------------------------------------------------------------------------

typedef __attribute__((ext_vector_type(16))) __bf16 v16bf;
typedef __attribute__((ext_vector_type(8)))  float  v8f;

#define D_MODEL 768
#define NHEAD   12
#define HEAD_DIM 64
#define SEQ     2048
#define BATCH   2
#define BROWS   (BATCH * SEQ)     // 4096 rows
#define QKV_N   (3 * D_MODEL)     // 2304
#define FFN_N   (4 * D_MODEL)     // 3072

#if defined(__HIP_DEVICE_COMPILE__) && \
    __has_builtin(__builtin_amdgcn_tensor_load_to_lds) && \
    __has_builtin(__builtin_amdgcn_s_wait_tensorcnt)
#define USE_TDM 1
#else
#define USE_TDM 0
#endif

static __device__ __forceinline__ float gelu_exact(float x) {
  return 0.5f * x * (1.0f + erff(x * 0.70710678118654752f));
}

#if USE_TDM
typedef __attribute__((ext_vector_type(4))) unsigned int u32x4;
typedef __attribute__((ext_vector_type(4))) int          i32x4;
typedef __attribute__((ext_vector_type(8))) int          i32x8;

// Issue one 2-D TDM tile load: global (row-major, row stride in elements)
// -> LDS with hardware padding (pad_interval/pad_amount codes per ISA 8.4).
// bf16 elements (data_size code 1 = 2 bytes). Whole-wave op; EXEC ignored.
static __device__ __forceinline__ void tdm_load_tile_2d(
    const void* gaddr, unsigned lds_addr,
    unsigned tile_w, unsigned tile_h, unsigned row_stride_elems,
    unsigned pad_interval_code, unsigned pad_amount_code) {
  const unsigned long long ga = (unsigned long long)gaddr;

  u32x4 g0;
  g0[0] = 1u;                                         // count=1, user descriptor
  g0[1] = lds_addr;                                   // LDS byte address
  g0[2] = (unsigned)(ga & 0xFFFFFFFFu);               // global_addr[31:0]
  g0[3] = (unsigned)((ga >> 32) & 0x01FFFFFFu)        // global_addr[56:32]
          | 0x80000000u;                              // type=2 ("image")

  i32x8 g1;
  g1[0] = (int)((1u << 16)                            // data_size = 2 bytes
              | (1u << 20)                            // pad_enable
              | (pad_interval_code << 22)
              | (pad_amount_code << 25));
  g1[1] = (int)((tile_w & 0xFFFFu) << 16);            // tensor_dim0[15:0]
  g1[2] = (int)(((tile_w >> 16) & 0xFFFFu)            // tensor_dim0[31:16]
              | ((tile_h & 0xFFFFu) << 16));          // tensor_dim1[15:0]
  g1[3] = (int)(((tile_h >> 16) & 0xFFFFu)            // tensor_dim1[31:16]
              | ((tile_w & 0xFFFFu) << 16));          // tile_dim0
  g1[4] = (int)(tile_h & 0xFFFFu);                    // tile_dim1 (tile_dim2=0)
  g1[5] = (int)row_stride_elems;                      // tensor_dim0_stride[31:0]
  g1[6] = 0;                                          // stride hi / dim1_stride lo
  g1[7] = 0;

  const i32x4 z4 = {0, 0, 0, 0};
#if __clang_major__ >= 23
  const i32x8 z8 = {0, 0, 0, 0, 0, 0, 0, 0};
  __builtin_amdgcn_tensor_load_to_lds(g0, g1, z4, z4, z8, 0);
#else
  __builtin_amdgcn_tensor_load_to_lds(g0, g1, z4, z4, 0);
#endif
}
#endif  // USE_TDM

// ---------------------------------------------------------------------------
// f32 -> bf16 conversion (weights)
// ---------------------------------------------------------------------------
__global__ __launch_bounds__(256) void tblk_cvt_bf16(const float* __restrict__ in,
                                                     __bf16* __restrict__ out,
                                                     size_t n) {
  size_t i = (size_t)blockIdx.x * blockDim.x + threadIdx.x;
  size_t stride = (size_t)gridDim.x * blockDim.x;
  for (; i < n; i += stride) out[i] = (__bf16)in[i];
}

// ---------------------------------------------------------------------------
// LayerNorm over rows of length C. Optional f32 and bf16 outputs.
// ---------------------------------------------------------------------------
__global__ __launch_bounds__(256) void tblk_layernorm(const float* __restrict__ in,
                                                      const float* __restrict__ g,
                                                      const float* __restrict__ b,
                                                      float* __restrict__ outF,
                                                      __bf16* __restrict__ outB,
                                                      int C) {
  __shared__ float red[256];
  const int row = blockIdx.x;
  const int tid = threadIdx.x;
  const float* xr = in + (size_t)row * C;

  float s = 0.f;
  for (int c = tid; c < C; c += 256) s += xr[c];
  red[tid] = s;
  __syncthreads();
  for (int off = 128; off > 0; off >>= 1) {
    if (tid < off) red[tid] += red[tid + off];
    __syncthreads();
  }
  const float mu = red[0] * (1.0f / (float)C);
  __syncthreads();

  float s2 = 0.f;
  for (int c = tid; c < C; c += 256) {
    float d = xr[c] - mu;
    s2 += d * d;
  }
  red[tid] = s2;
  __syncthreads();
  for (int off = 128; off > 0; off >>= 1) {
    if (tid < off) red[tid] += red[tid + off];
    __syncthreads();
  }
  const float var = red[0] * (1.0f / (float)C);
  const float inv = rsqrtf(var + 1e-5f);

  for (int c = tid; c < C; c += 256) {
    float v = (xr[c] - mu) * inv * g[c] + b[c];
    size_t idx = (size_t)row * C + c;
    if (outF) outF[idx] = v;
    if (outB) outB[idx] = (__bf16)v;
  }
}

// ---------------------------------------------------------------------------
// Tiled bf16 WMMA GEMM:  C[M,N] = A[M,K] @ B[K,N]  (+ epilogue)
//   mode 0: outB = bf16(acc + bias)
//   mode 1: outF = acc + bias
//   mode 2: outB = bf16(gelu(acc + bias))
//   mode 3: outF = outF + acc + bias        (residual add in-place)
// Block: 256 threads (8 waves). Tile 128x128, K-step 32.
// A staged by TDM (double-buffered); B staged transposed (Bt[n][k]) so both
// operand fragment reads are contiguous 32B (2 x ds_load_b128 per fragment).
// ---------------------------------------------------------------------------
__global__ __launch_bounds__(256) void tblk_gemm_bf16(const __bf16* __restrict__ A,
                                                      const __bf16* __restrict__ B,
                                                      const float* __restrict__ bias,
                                                      float* __restrict__ outF,
                                                      __bf16* __restrict__ outB,
                                                      int M, int N, int K, int mode) {
  __shared__ __bf16 As[2][128][40];   // 128 rows x 32 k (+8 pad) per buffer
  __shared__ __bf16 Bt[128][40];      // 128 n-cols x 32 k (+8 pad), transposed

  const int tid  = threadIdx.x;
  const int lane = tid & 31;
  const int wave = tid >> 5;
  const int lo = lane & 15;
  const int hi = lane >> 4;

  const int m0 = blockIdx.y * 128;
  const int n0 = blockIdx.x * 128;
  const int wm = (wave >> 1) * 32;   // 0,32,64,96
  const int wn = (wave & 1) * 64;    // 0,64

  v8f acc[2][4];
  const v8f vzero = {0.f, 0.f, 0.f, 0.f, 0.f, 0.f, 0.f, 0.f};
#pragma unroll
  for (int i = 0; i < 2; ++i)
#pragma unroll
    for (int j = 0; j < 4; ++j) acc[i][j] = vzero;

  int cur = 0;
  // B staging assignment: thread covers k-row (tid&31), 16-col chunk (tid>>5).
  const int rB = tid & 31;
  const int cB = (tid >> 5) * 16;

#if USE_TDM
  // Prologue: DMA first A K-tile into buffer 0. One wave issues; EXEC ignored.
  if (wave == 0) {
    // A tile: 32 elems wide x 128 rows, row stride K. 64B rows -> pad code 3,
    // pad 4 dwords (code 3) => LDS row stride 80B = 40 bf16.
    tdm_load_tile_2d(A + (size_t)m0 * K, (unsigned)(size_t)&As[0][0][0],
                     32u, 128u, (unsigned)K, 3u, 3u);
    __builtin_amdgcn_s_wait_tensorcnt(0);
  }
#else
  const int ra = tid >> 1;            // A row 0..127
  const int ca = (tid & 1) * 16;      // A col 0 or 16
#endif

  for (int k0 = 0; k0 < K; k0 += 32) {
#if USE_TDM
    // Prefetch next A K-tile into the other buffer while we compute this one.
    if (wave == 0 && (k0 + 32) < K) {
      tdm_load_tile_2d(A + (size_t)m0 * K + (k0 + 32),
                       (unsigned)(size_t)&As[cur ^ 1][0][0],
                       32u, 128u, (unsigned)K, 3u, 3u);
    }
#else
    {
      const uint4* src = reinterpret_cast<const uint4*>(A + (size_t)(m0 + ra) * K + k0 + ca);
      uint4 d0 = src[0];
      uint4 d1 = src[1];
      *reinterpret_cast<uint4*>(&As[0][ra][ca])     = d0;
      *reinterpret_cast<uint4*>(&As[0][ra][ca + 8]) = d1;
    }
#endif
    // Stage B tile transposed: read 16 contiguous bf16 of row k0+rB, scatter
    // into Bt so every operand column is k-contiguous.
    {
      const __bf16* src = B + (size_t)(k0 + rB) * N + n0 + cB;
      __bf16 tmp[16];
      *reinterpret_cast<uint4*>(&tmp[0]) = *reinterpret_cast<const uint4*>(src);
      *reinterpret_cast<uint4*>(&tmp[8]) = *reinterpret_cast<const uint4*>(src + 8);
#pragma unroll
      for (int e = 0; e < 16; ++e) Bt[cB + e][rB] = tmp[e];
    }
    __syncthreads();   // Bt (and staged A) visible to all waves

    // Build fragments in contiguous k order (k = j + 16*hi, same for A and B). Both reads contiguous.
    v16bf afrag[2];
#pragma unroll
    for (int i = 0; i < 2; ++i) {
      const int mrow = wm + i * 16 + lo;
#pragma unroll
      for (int j = 0; j < 16; ++j) {
        const int kk = j + 16 * hi;   // A: K = j + 16*hi
        afrag[i][j] = As[cur][mrow][kk];
      }
    }
    v16bf bfrag[4];
#pragma unroll
    for (int i = 0; i < 4; ++i) {
      const int ncol = wn + i * 16 + lo;
      const __bf16* p = &Bt[ncol][16 * hi];            // B: K = j + 16*hi, N = lo
#pragma unroll
      for (int j = 0; j < 16; ++j) bfrag[i][j] = p[j];
    }

#pragma unroll
    for (int i = 0; i < 2; ++i)
#pragma unroll
      for (int j = 0; j < 4; ++j)
        acc[i][j] = __builtin_amdgcn_wmma_f32_16x16x32_bf16(
            false, afrag[i], false, bfrag[j], (short)0, acc[i][j], false, false);

#if USE_TDM
    if (wave == 0) __builtin_amdgcn_s_wait_tensorcnt(0);  // prefetch landed
    __syncthreads();   // frag reads done: safe to overwrite Bt / swap As
    cur ^= 1;
#else
    __syncthreads();
#endif
  }

  // Epilogue. C/D layout: element r -> row = r + 8*hi, col lane = lo.
#pragma unroll
  for (int i = 0; i < 2; ++i) {
    const int mbase = m0 + wm + i * 16;
#pragma unroll
    for (int j = 0; j < 4; ++j) {
      const int nglob = n0 + wn + j * 16 + lo;
      const float bv = bias[nglob];
#pragma unroll
      for (int r = 0; r < 8; ++r) {
        const int row = mbase + r + 8 * hi;
        float v = acc[i][j][r] + bv;
        const size_t idx = (size_t)row * N + nglob;
        if (mode == 2) v = gelu_exact(v);
        if (mode == 3) v += outF[idx];
        if (mode == 0 || mode == 2) outB[idx] = (__bf16)v;
        else                        outF[idx] = v;
      }
    }
  }
}

// ---------------------------------------------------------------------------
// Flash-style causal attention. qkv: [BROWS, 2304] bf16, row layout
// [q | k | v], column = h*64 + d. Grid: (B*H, SEQ/128). Block 256 = 8 waves;
// each wave owns a 16-query tile; KV chunks of 32 with online softmax.
// ---------------------------------------------------------------------------
__global__ __launch_bounds__(256) void tblk_attention(const __bf16* __restrict__ qkv,
                                                      __bf16* __restrict__ ctx) {
  __shared__ __bf16 Ps[8][16][40];   // per-wave P staging (16x32 bf16, padded)

  const int tid  = threadIdx.x;
  const int lane = tid & 31;
  const int wave = tid >> 5;
  const int lo = lane & 15;
  const int hi = lane >> 4;

  const int bh = blockIdx.x;
  const int b  = bh / NHEAD;
  const int h  = bh % NHEAD;
  const int q0 = blockIdx.y * 128 + wave * 16;

  const size_t rstride = QKV_N;
  const __bf16* base = qkv + (size_t)b * SEQ * rstride;
  const int qoff = h * HEAD_DIM;
  const int koff = D_MODEL + h * HEAD_DIM;
  const int voff = 2 * D_MODEL + h * HEAD_DIM;

  // Q fragments: A layout, two K-halves of Dh=64.
  v16bf qf[2];
#pragma unroll
  for (int half = 0; half < 2; ++half) {
    const __bf16* qrow = base + (size_t)(q0 + lo) * rstride + qoff + 32 * half;
#pragma unroll
    for (int j = 0; j < 16; ++j) {
      const int d = j + 16 * hi;
      qf[half][j] = qrow[d];
    }
  }

  float mrow[8], lrow[8];
  v8f o[4];
  const v8f vzero = {0.f, 0.f, 0.f, 0.f, 0.f, 0.f, 0.f, 0.f};
#pragma unroll
  for (int r = 0; r < 8; ++r) { mrow[r] = -3.0e38f; lrow[r] = 0.f; }
#pragma unroll
  for (int dt = 0; dt < 4; ++dt) o[dt] = vzero;

  const int kvEnd = q0 + 16;                 // exclusive causal bound
  const float scale = 0.125f;                // 1/sqrt(64)

  for (int kv0 = 0; kv0 < kvEnd; kv0 += 32) {
    // --- scores S = Q @ K^T for 32 keys (two 16-key n-tiles) ---
    v8f s[2];
#pragma unroll
    for (int nt = 0; nt < 2; ++nt) {
      v16bf kf0, kf1;
      int key = kv0 + 16 * nt + lo;
      int keyc = key < SEQ - 1 ? key : SEQ - 1;
      const __bf16* krow = base + (size_t)keyc * rstride + koff;
#pragma unroll
      for (int j = 0; j < 16; ++j) {         // B: K(dim d) = j + 16*hi
        kf0[j] = krow[16 * hi + j];
        kf1[j] = krow[32 + 16 * hi + j];
      }
      v8f sv = __builtin_amdgcn_wmma_f32_16x16x32_bf16(
          false, qf[0], false, kf0, (short)0, vzero, false, false);
      s[nt] = __builtin_amdgcn_wmma_f32_16x16x32_bf16(
          false, qf[1], false, kf1, (short)0, sv, false, false);
    }

    // --- scale + causal mask + row max ---
    float pmax[8];
#pragma unroll
    for (int r = 0; r < 8; ++r) pmax[r] = -3.0e38f;
#pragma unroll
    for (int nt = 0; nt < 2; ++nt) {
      const int col = kv0 + 16 * nt + lo;
#pragma unroll
      for (int r = 0; r < 8; ++r) {
        const int row = q0 + r + 8 * hi;
        float v = s[nt][r] * scale;
        if (col > row) v = -3.0e38f;
        s[nt][r] = v;
        pmax[r] = fmaxf(pmax[r], v);
      }
    }
#pragma unroll
    for (int m = 1; m < 16; m <<= 1)
#pragma unroll
      for (int r = 0; r < 8; ++r)
        pmax[r] = fmaxf(pmax[r], __shfl_xor(pmax[r], m, 32));

    // --- online softmax update ---
    float alpha[8], rsum[8];
#pragma unroll
    for (int r = 0; r < 8; ++r) {
      const float mnew = fmaxf(mrow[r], pmax[r]);
      alpha[r] = __expf(mrow[r] - mnew);
      mrow[r] = mnew;
      lrow[r] *= alpha[r];
      rsum[r] = 0.f;
    }
#pragma unroll
    for (int dt = 0; dt < 4; ++dt)
#pragma unroll
      for (int r = 0; r < 8; ++r) o[dt][r] *= alpha[r];
#pragma unroll
    for (int nt = 0; nt < 2; ++nt)
#pragma unroll
      for (int r = 0; r < 8; ++r) {
        const float p = __expf(s[nt][r] - mrow[r]);
        s[nt][r] = p;
        rsum[r] += p;
      }
#pragma unroll
    for (int m = 1; m < 16; m <<= 1)
#pragma unroll
      for (int r = 0; r < 8; ++r) rsum[r] += __shfl_xor(rsum[r], m, 32);
#pragma unroll
    for (int r = 0; r < 8; ++r) lrow[r] += rsum[r];

    // --- stage P (C layout) -> A-fragment layout via per-wave LDS ---
#pragma unroll
    for (int nt = 0; nt < 2; ++nt)
#pragma unroll
      for (int r = 0; r < 8; ++r)
        Ps[wave][r + 8 * hi][16 * nt + lo] = (__bf16)s[nt][r];
    asm volatile("s_wait_dscnt 0" ::: "memory");   // same-wave LDS RAW

    v16bf pf;
#pragma unroll
    for (int j = 0; j < 16; ++j) {
      const int kk = j + 16 * hi;
      pf[j] = Ps[wave][lo][kk];
    }

    // --- O += P @ V (four 16-col d-tiles) ---
#pragma unroll
    for (int dt = 0; dt < 4; ++dt) {
      v16bf vf;
#pragma unroll
      for (int j = 0; j < 16; ++j) {
        int key = kv0 + j + 16 * hi;                // B: K(dim key) = j + 16*hi
        key = key < SEQ - 1 ? key : SEQ - 1;        // masked keys have p==0
        vf[j] = base[(size_t)key * rstride + voff + 16 * dt + lo];
      }
      o[dt] = __builtin_amdgcn_wmma_f32_16x16x32_bf16(
          false, pf, false, vf, (short)0, o[dt], false, false);
    }
  }

  // --- normalize and write ctx[B*L, 768] (col = h*64 + d) as bf16 ---
#pragma unroll
  for (int dt = 0; dt < 4; ++dt)
#pragma unroll
    for (int r = 0; r < 8; ++r) {
      const int row = q0 + r + 8 * hi;
      const float v = o[dt][r] / lrow[r];
      ctx[((size_t)b * SEQ + row) * D_MODEL + h * HEAD_DIM + 16 * dt + lo] = (__bf16)v;
    }
}

// ---------------------------------------------------------------------------
// Host-side orchestration
// ---------------------------------------------------------------------------
extern "C" void kernel_launch(void* const* d_in, const int* in_sizes, int n_in,
                              void* d_out, int out_size, void* d_ws, size_t ws_size,
                              hipStream_t stream) {
  (void)in_sizes; (void)n_in; (void)out_size; (void)ws_size;

  const float* x      = (const float*)d_in[0];
  const float* qkv_w  = (const float*)d_in[1];
  const float* qkv_b  = (const float*)d_in[2];
  const float* out_w  = (const float*)d_in[3];
  const float* out_b  = (const float*)d_in[4];
  const float* ln1_g  = (const float*)d_in[5];
  const float* ln1_b  = (const float*)d_in[6];
  const float* ln2_g  = (const float*)d_in[7];
  const float* ln2_b  = (const float*)d_in[8];
  const float* ffn_w1 = (const float*)d_in[9];
  const float* ffn_b1 = (const float*)d_in[10];
  const float* ffn_w2 = (const float*)d_in[11];
  const float* ffn_b2 = (const float*)d_in[12];
  float* outp = (float*)d_out;

  // Workspace carve-up (256B aligned).
  char* ws = (char*)d_ws;
  size_t off = 0;
  auto take = [&](size_t bytes) -> char* {
    char* p = ws + off;
    off += (bytes + 255) & ~(size_t)255;
    return p;
  };
  __bf16* wqkv   = (__bf16*)take((size_t)D_MODEL * QKV_N * 2);
  __bf16* wout   = (__bf16*)take((size_t)D_MODEL * D_MODEL * 2);
  __bf16* wff1   = (__bf16*)take((size_t)D_MODEL * FFN_N * 2);
  __bf16* wff2   = (__bf16*)take((size_t)FFN_N * D_MODEL * 2);
  __bf16* xn     = (__bf16*)take((size_t)BROWS * D_MODEL * 2);
  __bf16* qkvb   = (__bf16*)take((size_t)BROWS * QKV_N * 2);
  __bf16* ctx    = (__bf16*)take((size_t)BROWS * D_MODEL * 2);
  float*  attno  = (float*) take((size_t)BROWS * D_MODEL * 4);
  __bf16* ln2b   = (__bf16*)take((size_t)BROWS * D_MODEL * 2);
  __bf16* hbuf   = (__bf16*)take((size_t)BROWS * FFN_N * 2);

  // 1) weights -> bf16
  tblk_cvt_bf16<<<dim3(1024), 256, 0, stream>>>(qkv_w,  wqkv, (size_t)D_MODEL * QKV_N);
  tblk_cvt_bf16<<<dim3(1024), 256, 0, stream>>>(out_w,  wout, (size_t)D_MODEL * D_MODEL);
  tblk_cvt_bf16<<<dim3(1024), 256, 0, stream>>>(ffn_w1, wff1, (size_t)D_MODEL * FFN_N);
  tblk_cvt_bf16<<<dim3(1024), 256, 0, stream>>>(ffn_w2, wff2, (size_t)FFN_N * D_MODEL);

  // 2) LN1 -> bf16 activations
  tblk_layernorm<<<dim3(BROWS), 256, 0, stream>>>(x, ln1_g, ln1_b, nullptr, xn, D_MODEL);

  // 3) QKV projection: [4096,768] @ [768,2304] -> bf16 qkv (+bias)
  tblk_gemm_bf16<<<dim3(QKV_N / 128, BROWS / 128), 256, 0, stream>>>(
      xn, wqkv, qkv_b, nullptr, qkvb, BROWS, QKV_N, D_MODEL, 0);

  // 4) causal attention -> ctx bf16
  tblk_attention<<<dim3(BATCH * NHEAD, SEQ / 128), 256, 0, stream>>>(qkvb, ctx);

  // 5) out projection -> f32 (+bias)
  tblk_gemm_bf16<<<dim3(D_MODEL / 128, BROWS / 128), 256, 0, stream>>>(
      ctx, wout, out_b, attno, nullptr, BROWS, D_MODEL, D_MODEL, 1);

  // 6) LN2 -> d_out (residual base, f32) + bf16 copy for FFN
  tblk_layernorm<<<dim3(BROWS), 256, 0, stream>>>(attno, ln2_g, ln2_b, outp, ln2b, D_MODEL);

  // 7) FFN1 + exact GELU -> bf16
  tblk_gemm_bf16<<<dim3(FFN_N / 128, BROWS / 128), 256, 0, stream>>>(
      ln2b, wff1, ffn_b1, nullptr, hbuf, BROWS, FFN_N, D_MODEL, 2);

  // 8) FFN2 + bias + residual add into d_out (f32)
  tblk_gemm_bf16<<<dim3(D_MODEL / 128, BROWS / 128), 256, 0, stream>>>(
      hbuf, wff2, ffn_b2, outp, nullptr, BROWS, D_MODEL, FFN_N, 3);
}
